// BatchWiseSimCLR_35055523070222
// MI455X (gfx1250) — matrix operand order repaired and checked by reading, not compiled, not run on hardware
//
#include <hip/hip_runtime.h>
#include <hip/hip_bf16.h>

// Problem constants (from reference): B=64, 2B=128, K=256, MS=512, D=256, TEMP=0.5
#define TWO_B 128
#define KCH   256
#define MS    512
#define DIM   256
#define NROWS 32768          // TWO_B * KCH
#define EPS   1e-8f

typedef __attribute__((ext_vector_type(16))) __bf16 v16bf;
typedef __attribute__((ext_vector_type(8)))  float  v8f;
typedef int i32x4 __attribute__((vector_size(16)));   // matches builtin param type

// Address-space casts for CDNA5 async-to-LDS ops. Low 32 bits of a generic
// shared-aperture address are the LDS byte offset; global generic == AS(1).
#define TO_AS1_V4(p) ((__attribute__((address_space(1))) i32x4*)(uintptr_t)(p))
#define TO_AS3_V4(p) ((__attribute__((address_space(3))) i32x4*)(uint32_t)(uintptr_t)(p))

// ---------------------------------------------------------------------------
// Init: zero the two accumulators (accum[0] = pos sum, accum[1] = neg lse sum)
// ---------------------------------------------------------------------------
__global__ void simclr_init_kernel(float* accum) {
    accum[0] = 0.0f;
    accum[1] = 0.0f;
}

// ---------------------------------------------------------------------------
// Phase 1: per-row norms, positive-loss accumulation, write normalized bf16
// z into workspace, layout z[b][i][d]  (b=0..127, i=0..511, d=0..255).
// Row R of x maps to (b = R & 127, i = R >> 7); x_pair maps to i + 256.
// Grid: 32768 blocks x 256 threads (one block per row pair).
// ---------------------------------------------------------------------------
__global__ void simclr_prep_kernel(const float* __restrict__ x,
                                   const float* __restrict__ xp,
                                   __hip_bfloat16* __restrict__ zbf,
                                   float* __restrict__ accum) {
    __shared__ float red[3][8];
    __shared__ float bc[2];

    const int g   = blockIdx.x;       // row index 0..32767
    const int tid = threadIdx.x;      // 0..255 == feature index d
    const size_t base = (size_t)g * DIM + tid;

    const float xv = x[base];
    const float pv = xp[base];
    float sx = xv * xv, sp = pv * pv, sd = xv * pv;

    // wave32 reduction
    #pragma unroll
    for (int o = 16; o > 0; o >>= 1) {
        sx += __shfl_xor(sx, o);
        sp += __shfl_xor(sp, o);
        sd += __shfl_xor(sd, o);
    }
    const int wave = tid >> 5, lane = tid & 31;
    if (lane == 0) { red[0][wave] = sx; red[1][wave] = sp; red[2][wave] = sd; }
    __syncthreads();

    if (tid == 0) {
        float tx = 0.f, tp = 0.f, td = 0.f;
        #pragma unroll
        for (int w = 0; w < 8; ++w) { tx += red[0][w]; tp += red[1][w]; td += red[2][w]; }
        const float nx = fmaxf(sqrtf(tx), EPS);
        const float np = fmaxf(sqrtf(tp), EPS);
        bc[0] = 1.0f / nx;
        bc[1] = 1.0f / np;
        // positive loss term: cos/TEMP  (n_positives == 1)
        atomicAdd(&accum[0], (td / (nx * np)) * 2.0f);
    }
    __syncthreads();

    const float invx = bc[0], invp = bc[1];
    const int i  = g >> 7;
    const int bb = g & 127;
    const size_t zx = ((size_t)bb * MS + i)        * DIM + tid;
    const size_t zp = ((size_t)bb * MS + KCH + i)  * DIM + tid;
    zbf[zx] = __float2bfloat16(xv * invx);
    zbf[zp] = __float2bfloat16(pv * invp);
}

// ---------------------------------------------------------------------------
// Phase 2: per batch b, Gram(Z_b) with bf16 WMMA from LDS, fixed-shift
// logsumexp (rows are unit vectors so sim in [-2,2]; shift by the exact max 2).
// Grid: 128 blocks x 512 threads (16 waves).
// Dynamic LDS: 512*256 bf16 = 256 KB (+16 B scratch) of the 320 KB WGP pool.
// Staging uses CDNA5 async global->LDS (ASYNCcnt) instead of VGPR round-trips.
// ---------------------------------------------------------------------------
extern __shared__ char smem_raw[];

__global__ void simclr_neg_kernel(const __hip_bfloat16* __restrict__ zbf,
                                  float* __restrict__ accum) {
    __hip_bfloat16* z = reinterpret_cast<__hip_bfloat16*>(smem_raw);
    float* lacc = reinterpret_cast<float*>(smem_raw + (size_t)MS * DIM * 2);

    const int tid = threadIdx.x;
    const int b   = blockIdx.x;
    if (tid == 0) *lacc = 0.0f;

    // ---- stage the 512x256 bf16 batch tile into LDS with async b128 copies
    {
        const __hip_bfloat16* src = zbf + (size_t)b * MS * DIM;
        const int n16 = (MS * DIM * 2) / 16;   // 16384 x 16-byte packets
        for (int i = tid; i < n16; i += 512) {
#if __has_builtin(__builtin_amdgcn_global_load_async_to_lds_b128)
            __builtin_amdgcn_global_load_async_to_lds_b128(
                TO_AS1_V4((const char*)src + (size_t)i * 16),
                TO_AS3_V4((char*)z + (size_t)i * 16), 0, 0);
#else
            uint32_t lds_off = (uint32_t)(uintptr_t)((char*)z + (size_t)i * 16);
            uint64_t gaddr   = (uint64_t)(uintptr_t)((const char*)src + (size_t)i * 16);
            asm volatile("global_load_async_to_lds_b128 %0, %1, off"
                         :: "v"(lds_off), "v"(gaddr) : "memory");
#endif
        }
#if __has_builtin(__builtin_amdgcn_s_wait_asynccnt)
        __builtin_amdgcn_s_wait_asynccnt(0);
#else
        asm volatile("s_wait_asynccnt 0x0" ::: "memory");
#endif
    }
    __syncthreads();

    const int wave = tid >> 5;
    const int lane = tid & 31;
    const int half = lane >> 4;     // 0 or 1
    const int l16  = lane & 15;

    #pragma unroll
    for (int tt = 0; tt < 2; ++tt) {
        const int t    = wave + tt * 16;      // tile row 0..31
        const int rowA = t * 16 + l16;

        // A operand k order chosen here (not the ISA table order): lane half 0 holds K=k0..k0+15
        // contiguously, half 1 holds K=k0+16..k0+31 (matches B)
        v16bf a[8];
        #pragma unroll
        for (int kc = 0; kc < 8; ++kc) {
            const int k0 = kc * 32 + half * 16;
            union { uint4 u[2]; v16bf v; } ta;
            ta.u[0] = *reinterpret_cast<const uint4*>(z + rowA * DIM + k0);
            ta.u[1] = *reinterpret_cast<const uint4*>(z + rowA * DIM + k0 + 8);
            a[kc] = ta.v;
        }

        float acc[8] = {0.f, 0.f, 0.f, 0.f, 0.f, 0.f, 0.f, 0.f};

        for (int jt = 0; jt < 32; ++jt) {
            const int rowB = jt * 16 + l16;
            // B operand k order chosen here (same as A): half 0 holds K=k0..k0+15
            // contiguously, half 1 holds K=k0+16..k0+31. In b128 units the
            // chunk for kstep kc starts at uint4 index kc*4 + half*2.
            const uint4* bp = reinterpret_cast<const uint4*>(z + (size_t)rowB * DIM);

            union BU { uint4 u[2]; v16bf v; };
            BU buf[2];
            buf[0].u[0] = bp[half * 2 + 0];
            buf[0].u[1] = bp[half * 2 + 1];

            v8f c = {};
            #pragma unroll
            for (int kc = 0; kc < 8; ++kc) {
                if (kc < 7) {   // software pipeline: prefetch next B chunk
                    buf[(kc + 1) & 1].u[0] = bp[(kc + 1) * 4 + half * 2 + 0];
                    buf[(kc + 1) & 1].u[1] = bp[(kc + 1) * 4 + half * 2 + 1];
                }
                c = __builtin_amdgcn_wmma_f32_16x16x32_bf16(
                        false, a[kc], false, buf[kc & 1].v, (short)0, c, false, false);
            }
            // C/D layout: VGPR r, lanes 0-15 -> (M=r, N=lane);
            //             lanes 16-31 -> (M=r+8, N=lane-16)
            const bool diag_tile = (jt == t);
            #pragma unroll
            for (int r = 0; r < 8; ++r) {
                float e = __expf(2.0f * c[r] - 2.0f);   // exp(sim - 2)
                if (diag_tile && (lane == r || lane == r + 24)) e = 0.0f;
                acc[r] += e;
            }
        }

        // reduce each row's exp-sum over its 16 lanes, then lse = 2 + log(s)
        #pragma unroll
        for (int r = 0; r < 8; ++r) {
            float s = acc[r];
            s += __shfl_xor(s, 1);
            s += __shfl_xor(s, 2);
            s += __shfl_xor(s, 4);
            s += __shfl_xor(s, 8);
            if (l16 == 0) atomicAdd(lacc, 2.0f + __logf(s));
        }
    }
    __syncthreads();
    if (tid == 0) atomicAdd(&accum[1], *lacc);
}

// ---------------------------------------------------------------------------
// Finalize: loss = mean(lse) - mean(cos/TEMP)
// ---------------------------------------------------------------------------
__global__ void simclr_final_kernel(const float* accum, float* out) {
    out[0] = accum[1] / (float)(TWO_B * MS) - accum[0] / (float)NROWS;
}

// ---------------------------------------------------------------------------
extern "C" void kernel_launch(void* const* d_in, const int* in_sizes, int n_in,
                              void* d_out, int out_size, void* d_ws, size_t ws_size,
                              hipStream_t stream) {
    const float* x  = (const float*)d_in[0];
    const float* xp = (const float*)d_in[1];
    float* out = (float*)d_out;

    // workspace layout: [0, 32MB) = normalized bf16 z[b][i][d]; then 2 floats
    __hip_bfloat16* zbf = (__hip_bfloat16*)d_ws;
    float* accum = (float*)((char*)d_ws + (size_t)TWO_B * MS * DIM * 2);

    simclr_init_kernel<<<1, 1, 0, stream>>>(accum);
    simclr_prep_kernel<<<NROWS, 256, 0, stream>>>(x, xp, zbf, accum);

    const size_t lds_bytes = (size_t)MS * DIM * 2 + 16;   // 256 KB + scratch
    (void)hipFuncSetAttribute(reinterpret_cast<const void*>(simclr_neg_kernel),
                              hipFuncAttributeMaxDynamicSharedMemorySize,
                              (int)lds_bytes);
    simclr_neg_kernel<<<TWO_B, 512, lds_bytes, stream>>>(zbf, accum);

    simclr_final_kernel<<<1, 1, 0, stream>>>(accum, out);
}
